// JastrowFactor_80229989089986
// MI455X (gfx1250) — matrix operand order repaired and checked by reading, not compiled, not run on hardware
//
#include <hip/hip_runtime.h>
#include <stdint.h>

// MI455X / gfx1250: wave32, WMMA f32 = f16 A/B with f32 accumulate.
typedef __attribute__((ext_vector_type(16))) _Float16 v16h;
typedef __attribute__((ext_vector_type(8)))  float    v8f;

#define N_WALK 8192
#define N_E    32
#define N_NUC  8
#define D_H    32
#define N_PAIR 496   // 31*16
#define N_TILES 33   // 31 ee tiles + 2 en tiles

union V16H { v16h v; _Float16 e[16]; };
union V8F  { v8f  v; float    e[8];  };

// Fast silu: one v_exp_f32 + one v_rcp_f32 (both TRANS, co-execute with VALU)
__device__ __forceinline__ float silu_f(float x) {
    return x * __builtin_amdgcn_rcpf(1.0f + __expf(-x));
}
// Fast Pade r/(1+b*r)
__device__ __forceinline__ float pade_f(float r, float b) {
    return r * __builtin_amdgcn_rcpf(fmaf(b, r, 1.0f));
}

// layer-2 bias + silu + layer-3 dot over this lane's two columns, this lane's 8 rows summed
__device__ __forceinline__ float tile_tail(const V8F& clo, const V8F& chi,
                                           float b2lo, float b2hi,
                                           float w3lo, float w3hi) {
    float nn = 0.0f;
    #pragma unroll
    for (int rr = 0; rr < 8; ++rr) {
        nn = fmaf(silu_f(clo.e[rr] + b2lo), w3lo, nn);
        nn = fmaf(silu_f(chi.e[rr] + b2hi), w3hi, nn);
    }
    return nn;
}

__global__ __launch_bounds__(256) void jastrow_kernel(
    const float* __restrict__ r_e,       // [W,32,3]
    const float* __restrict__ r_nuc,     // [8,3]
    const float* __restrict__ charges,   // [8]
    const uint8_t* __restrict__ spin,    // [32,32] bool (1 byte each)
    const float* __restrict__ b_en,      // [8]
    const float* __restrict__ b_ee,      // [1]
    const float* __restrict__ W1_en, const float* __restrict__ b1_en,
    const float* __restrict__ W2_en, const float* __restrict__ b2_en,
    const float* __restrict__ W3_en, const float* __restrict__ b3_en,
    const float* __restrict__ W1_ee, const float* __restrict__ b1_ee,
    const float* __restrict__ W2_ee, const float* __restrict__ b2_ee,
    const float* __restrict__ W3_ee, const float* __restrict__ b3_ee,
    const float* __restrict__ scale_en, const float* __restrict__ scale_ee,
    float* __restrict__ out)             // [W]
{
    __shared__ float sW2ee[D_H * D_H];
    __shared__ float sW2en[D_H * D_H];
    __shared__ float sW1en[N_NUC * D_H];
    __shared__ float sRe[N_E * 3];
    __shared__ float sPairR[N_PAIR];       // e-e distances
    __shared__ float sX2[N_E * N_NUC];     // e-n squared distances
    __shared__ float sRed[8];

    const int w   = blockIdx.x;
    const int tid = threadIdx.x;

    // ---- Phase 1: stage weights + this walker's electron positions into LDS ----
    for (int i = tid; i < D_H * D_H; i += 256) {
        sW2ee[i] = W2_ee[i];
        sW2en[i] = W2_en[i];
    }
    if (tid < N_NUC * D_H) sW1en[tid] = W1_en[tid];
    for (int i = tid; i < N_E * 3; i += 256) sRe[i] = r_e[(size_t)w * (N_E * 3) + i];
    __syncthreads();

    const float bee_s = logf(1.0f + __expf(b_ee[0]));
    const float sc_en = scale_en[0], sc_ee = scale_ee[0];
    const float b3ee  = b3_ee[0],    b3en  = b3_en[0];

    // ---- Phase 2: geometry + analytic terms, once per block (not in hot loop) ----
    float acc = 0.0f;   // per-thread partial of this walker's J (reduced at end)
    for (int p = tid; p < N_PAIR; p += 256) {
        // triangular (i<j) indices from flat p: c(i) = i*(63-i)/2
        int i = (int)floorf((63.0f - __builtin_amdgcn_sqrtf(3969.0f - 8.0f * (float)p)) * 0.5f);
        if (i < 0) i = 0;
        while ((i * (63 - i)) / 2 > p) --i;
        while (((i + 1) * (62 - i)) / 2 <= p) ++i;
        int j = p - (i * (63 - i)) / 2 + i + 1;

        float dx = sRe[i*3+0] - sRe[j*3+0];
        float dy = sRe[i*3+1] - sRe[j*3+1];
        float dz = sRe[i*3+2] - sRe[j*3+2];
        float r  = __builtin_amdgcn_sqrtf(fmaf(dx, dx, fmaf(dy, dy, dz*dz)));
        sPairR[p] = r;
        float aspin = spin[i * N_E + j] ? 0.25f : 0.5f;
        acc += aspin * pade_f(r, bee_s);
    }
    {   // one (electron, nucleus) cell per thread: 32*8 == 256
        int e = tid >> 3, k = tid & 7;
        float dx = sRe[e*3+0] - r_nuc[k*3+0];
        float dy = sRe[e*3+1] - r_nuc[k*3+1];
        float dz = sRe[e*3+2] - r_nuc[k*3+2];
        float r2 = fmaf(dx, dx, fmaf(dy, dy, dz*dz));
        sX2[e * N_NUC + k] = r2;
        float bs = logf(1.0f + __expf(b_en[k]));
        acc = fmaf(-charges[k], pade_f(__builtin_amdgcn_sqrtf(r2), bs), acc);
    }
    __syncthreads();

    const int wave  = tid >> 5;
    const int lane  = tid & 31;
    const int m     = lane & 15;     // tile row handled by this lane (A layout)
    const bool hiK  = lane >= 16;    // lanes 16-31 carry the high-K halves
    const int n     = lane & 15;     // D/B column owned by this lane (lo tile)
    const int kbase = hiK ? 16 : 0;  // unused; B layout below starts hi lanes at K=8

    // ---- Hoisted, loop-invariant operands ----
    // All four B tiles in registers. B layout: slot s <-> K = s + (s<8?0:8) + (hiK?8:0), col N = n (+16 hi).
    V16H BloEE, BhiEE, BloEN, BhiEN;
    #pragma unroll
    for (int s = 0; s < 16; ++s) {
        int kk = s + (s < 8 ? 0 : 8) + (hiK ? 8 : 0);
        BloEE.e[s] = (_Float16)sW2ee[kk * D_H + n];
        BhiEE.e[s] = (_Float16)sW2ee[kk * D_H + n + 16];
        BloEN.e[s] = (_Float16)sW2en[kk * D_H + n];
        BhiEN.e[s] = (_Float16)sW2en[kk * D_H + n + 16];
    }
    // ee layer-1 params for this lane's 16 A slots: K index jj(s)
    float w1ee[16], b1ee[16];
    #pragma unroll
    for (int s = 0; s < 16; ++s) {
        int jj = s + (s < 8 ? 0 : 8) + (hiK ? 8 : 0);
        w1ee[s] = W1_ee[jj];
        b1ee[s] = b1_ee[jj];
    }
    // epilogue scalars per type at this lane's columns
    const float b2loEE = b2_ee[n], b2hiEE = b2_ee[n + 16];
    const float w3loEE = W3_ee[n], w3hiEE = W3_ee[n + 16];
    const float b2loEN = b2_en[n], b2hiEN = b2_en[n + 16];
    const float w3loEN = W3_en[n], w3hiEN = W3_en[n + 16];

    // ---- Phase 3: pure NN tile loop (uniform per wave -> EXEC all-ones) ----
    for (int t = wave; t < N_TILES; t += 8) {
        V16H A;
        if (t < 31) {
            // electron-electron tile: 16 pairs; r precomputed
            float r = sPairR[t * 16 + m];
            #pragma unroll
            for (int s = 0; s < 16; ++s)
                A.e[s] = (_Float16)silu_f(fmaf(r, w1ee[s], b1ee[s]));

            V8F Clo, Chi;
            #pragma unroll
            for (int q = 0; q < 8; ++q) { Clo.e[q] = 0.0f; Chi.e[q] = 0.0f; }
            Clo.v = __builtin_amdgcn_wmma_f32_16x16x32_f16(false, A.v, false, BloEE.v,
                                                           (short)0, Clo.v, false, false);
            Chi.v = __builtin_amdgcn_wmma_f32_16x16x32_f16(false, A.v, false, BhiEE.v,
                                                           (short)0, Chi.v, false, false);
            float nn = tile_tail(Clo, Chi, b2loEE, b2hiEE, w3loEE, w3hiEE);
            if (lane == 0) nn = fmaf(16.0f, b3ee, nn);   // +b3 per row, 16 rows/tile
            acc = fmaf(sc_ee, nn, acc);
        } else {
            // electron-nucleus tile: 16 electrons; x2 precomputed
            int e = (t - 31) * 16 + m;
            float x2[N_NUC];
            #pragma unroll
            for (int k = 0; k < N_NUC; ++k) x2[k] = sX2[e * N_NUC + k];
            #pragma unroll
            for (int s = 0; s < 16; ++s) {
                int jj = s + (s < 8 ? 0 : 8) + (hiK ? 8 : 0);
                float z = b1_en[jj];
                #pragma unroll
                for (int k = 0; k < N_NUC; ++k) z = fmaf(x2[k], sW1en[k*D_H + jj], z);
                A.e[s] = (_Float16)silu_f(z);
            }
            V8F Clo, Chi;
            #pragma unroll
            for (int q = 0; q < 8; ++q) { Clo.e[q] = 0.0f; Chi.e[q] = 0.0f; }
            Clo.v = __builtin_amdgcn_wmma_f32_16x16x32_f16(false, A.v, false, BloEN.v,
                                                           (short)0, Clo.v, false, false);
            Chi.v = __builtin_amdgcn_wmma_f32_16x16x32_f16(false, A.v, false, BhiEN.v,
                                                           (short)0, Chi.v, false, false);
            float nn = tile_tail(Clo, Chi, b2loEN, b2hiEN, w3loEN, w3hiEN);
            if (lane == 0) nn = fmaf(16.0f, b3en, nn);
            acc = fmaf(sc_en, nn, acc);
        }
    }

    // wave32 reduction, then cross-wave via LDS, single store per walker
    #pragma unroll
    for (int off = 16; off > 0; off >>= 1)
        acc += __shfl_xor(acc, off, 32);
    if (lane == 0) sRed[wave] = acc;
    __syncthreads();
    if (tid == 0) {
        float s = 0.0f;
        #pragma unroll
        for (int q = 0; q < 8; ++q) s += sRed[q];
        out[w] = s;
    }
}

extern "C" void kernel_launch(void* const* d_in, const int* in_sizes, int n_in,
                              void* d_out, int out_size, void* d_ws, size_t ws_size,
                              hipStream_t stream) {
    (void)in_sizes; (void)n_in; (void)out_size; (void)d_ws; (void)ws_size;
    jastrow_kernel<<<N_WALK, 256, 0, stream>>>(
        (const float*)d_in[0],   // r_electrons
        (const float*)d_in[1],   // r_nuclei
        (const float*)d_in[2],   // charges
        (const uint8_t*)d_in[3], // spin_mask_parallel (bool)
        (const float*)d_in[4],   // b_en
        (const float*)d_in[5],   // b_ee
        (const float*)d_in[6],  (const float*)d_in[7],   // W1_en, b1_en
        (const float*)d_in[8],  (const float*)d_in[9],   // W2_en, b2_en
        (const float*)d_in[10], (const float*)d_in[11],  // W3_en, b3_en
        (const float*)d_in[12], (const float*)d_in[13],  // W1_ee, b1_ee
        (const float*)d_in[14], (const float*)d_in[15],  // W2_ee, b2_ee
        (const float*)d_in[16], (const float*)d_in[17],  // W3_ee, b3_ee
        (const float*)d_in[18], (const float*)d_in[19],  // scale_en, scale_ee
        (float*)d_out);
}
